// ScaledDotProductAttention_15453292331079
// MI455X (gfx1250) — matrix operand order repaired and checked by reading, not compiled, not run on hardware
//
#include <hip/hip_runtime.h>
#include <hip/hip_bf16.h>
#include <cstdint>
#include <cstddef>

// Problem constants (match reference)
constexpr int BATCH = 16;
constexpr int SLQ   = 2048;
constexpr int SLK   = 2048;
constexpr int DIM   = 512;   // DQ == DK == DATTN == DV
constexpr float INV_TEMP = 0.04419417382415922f;  // 1/sqrt(512)

typedef __attribute__((ext_vector_type(16))) _Float16 v16h;
typedef __attribute__((ext_vector_type(2)))  _Float16 h2;
typedef __attribute__((ext_vector_type(8)))  float    v8f;

#define WMMA_F16(a, b, c) \
    __builtin_amdgcn_wmma_f32_16x16x32_f16(false, (a), false, (b), (short)0, (c), false, false)

// ---------------------------------------------------------------------------
// CDNA5 async global->LDS copy (ASYNCcnt-tracked; CDNA5 ISA).
// GVS mode: 64-bit SGPR base + per-lane 32-bit byte offset; VDST holds the
// per-lane LDS byte address (low 32 bits of the generic shared pointer).
// ---------------------------------------------------------------------------
__device__ __forceinline__ uint32_t lds_addr32(const void* p) {
    return (uint32_t)(size_t)p;   // ISA: LDS addr = generic addr[31:0]
}

__device__ __forceinline__ void async_ld_b128(uint32_t lds_byte, const void* sbase,
                                              uint32_t gbyte_off) {
    asm volatile("global_load_async_to_lds_b128 %0, %1, %2"
                 :
                 : "v"(lds_byte), "v"(gbyte_off), "s"(sbase)
                 : "memory");
}

__device__ __forceinline__ void wait_asynccnt0() {
    asm volatile("s_wait_asynccnt 0x0" ::: "memory");
}

// ---------------------------------------------------------------------------
// WMMA fragment loaders (wave32; contiguous k = 16*(lane/16) + s in both A and B)
// ---------------------------------------------------------------------------

// A-matrix 16x32 (MxK), f16, from LDS tile with row stride `ld`.
__device__ __forceinline__ v16h frag_a(const _Float16* base, int ld) {
    const int l = threadIdx.x & 31;
    const _Float16* rp = base + (l & 15) * ld + ((l >> 4) << 4);
    v16h a;
#pragma unroll
    for (int r = 0; r < 8; ++r) a[r] = rp[r];          // K = ko+0 .. ko+7
#pragma unroll
    for (int r = 0; r < 8; ++r) a[8 + r] = rp[8 + r];  // K = ko+8 .. ko+15
    return a;
}

// B-matrix 32x16 (KxN) from an LDS tile stored TRANSPOSED as Bs[n][k], row stride `ld`.
__device__ __forceinline__ v16h frag_b_trans(const _Float16* base, int ld) {
    const int l = threadIdx.x & 31;
    const _Float16* rp = base + (l & 15) * ld + ((l >> 4) << 4);
    v16h b;
#pragma unroll
    for (int r = 0; r < 16; ++r) b[r] = rp[r];
    return b;
}

// B-matrix 32x16 (KxN) from an LDS tile stored DIRECT as Bs[k][n], row stride `ld`.
__device__ __forceinline__ v16h frag_b_direct(const _Float16* base, int ld) {
    const int l = threadIdx.x & 31;
    const int n  = l & 15;
    const int kb = (l >> 4) << 4;
    v16h b;
#pragma unroll
    for (int r = 0; r < 16; ++r) b[r] = base[(kb + r) * ld + n];
    return b;
}

// Pack two adjacent f32 into an f16 pair (v_cvt_pk_f16_f32 + single b32 LDS store).
__device__ __forceinline__ h2 cvt_pk(const float* src) {
    float2 f = *(const float2*)src;
    h2 o;
    o[0] = (_Float16)f.x;
    o[1] = (_Float16)f.y;
    return o;
}

// ---------------------------------------------------------------------------
// Kernel 1: projection GEMM  Y[m,n] = f16( sum_k X[m,k]*W[n,k] + bias[n] )
// Block tile 128x64, BK=32, 8 waves -> 4x2 wave grid, 32x32 per wave.
// ---------------------------------------------------------------------------
__global__ __launch_bounds__(256) void proj_kernel(const float* __restrict__ X,
                                                   const float* __restrict__ W,
                                                   const float* __restrict__ bias,
                                                   _Float16* __restrict__ Y) {
    __shared__ _Float16 As[128 * 32];   // [m][k]
    __shared__ _Float16 Bs[64 * 32];    // [n][k]  (W rows are already [n][k])
    const int tid = threadIdx.x;
    const int wid = tid >> 5;
    const int wm = wid >> 1, wn = wid & 1;
    const int m0 = blockIdx.x * 128;
    const int n0 = blockIdx.y * 64;

    v8f c[2][2] = {};

    for (int k0 = 0; k0 < DIM; k0 += 32) {
#pragma unroll
        for (int i = 0; i < 8; ++i) {              // 2048 f16-pairs, f32->f16
            int u = tid + 256 * i;                 // pair index; 16 pairs/row
            int r = u >> 4, c2 = (u & 15) * 2;
            ((h2*)As)[u] = cvt_pk(&X[(size_t)(m0 + r) * DIM + k0 + c2]);
        }
#pragma unroll
        for (int i = 0; i < 4; ++i) {              // 1024 f16-pairs
            int u = tid + 256 * i;
            int r = u >> 4, c2 = (u & 15) * 2;
            ((h2*)Bs)[u] = cvt_pk(&W[(size_t)(n0 + r) * DIM + k0 + c2]);
        }
        __syncthreads();

        v16h a0 = frag_a(As + (wm * 32) * 32, 32);
        v16h a1 = frag_a(As + (wm * 32 + 16) * 32, 32);
        v16h b0 = frag_b_trans(Bs + (wn * 32) * 32, 32);
        v16h b1 = frag_b_trans(Bs + (wn * 32 + 16) * 32, 32);
        c[0][0] = WMMA_F16(a0, b0, c[0][0]);
        c[0][1] = WMMA_F16(a0, b1, c[0][1]);
        c[1][0] = WMMA_F16(a1, b0, c[1][0]);
        c[1][1] = WMMA_F16(a1, b1, c[1][1]);
        __syncthreads();
    }

    const int l  = tid & 31;
    const int ln = l & 15;
    const int mb = (l >> 4) << 3;
#pragma unroll
    for (int im = 0; im < 2; ++im)
#pragma unroll
        for (int in_ = 0; in_ < 2; ++in_)
#pragma unroll
            for (int r = 0; r < 8; ++r) {
                int m = m0 + wm * 32 + im * 16 + mb + r;
                int n = n0 + wn * 32 + in_ * 16 + ln;
                float val = c[im][in_][r] + bias[n];
                Y[(size_t)m * DIM + n] = (_Float16)val;
            }
}

// ---------------------------------------------------------------------------
// Kernel 2: scores  S[b,i,j] = mask ? -inf : (qp[b,i,:] . kp[b,j,:]) / temp
// Block tile 128x128, wave tile 32x64 (8 WMMAs/K-step). qp/kp already f16 ->
// double-buffered GLOBAL_LOAD_ASYNC_TO_LDS_B128 staging (ASYNCcnt pipeline):
// tile k+1 is in flight while tile k is consumed by the WMMAs.
// ---------------------------------------------------------------------------
__global__ __launch_bounds__(256) void scores_kernel(const _Float16* __restrict__ QP,
                                                     const _Float16* __restrict__ KP,
                                                     const unsigned char* __restrict__ mask,
                                                     float* __restrict__ S) {
    __shared__ _Float16 As[2][128 * 32];   // [i][k], double buffered (8KB each)
    __shared__ _Float16 Bs[2][128 * 32];   // [j][k], double buffered (8KB each)
    const int tid = threadIdx.x;
    const int wid = tid >> 5;
    const int wm = wid >> 1, wn = wid & 1;     // 4x2 wave grid; wave tile 32x64
    const int m0 = blockIdx.x * 128;
    const int n0 = blockIdx.y * 128;
    const int b  = blockIdx.z;
    const _Float16* A  = QP + (size_t)b * SLQ * DIM;
    const _Float16* Bp = KP + (size_t)b * SLK * DIM;

    const uint32_t asA = lds_addr32(&As[0][0]);
    const uint32_t asB = lds_addr32(&Bs[0][0]);

    // Issue one 32-wide K-slice of both tiles into LDS buffer `buf`.
    auto issue_tile = [&](int k0, int buf) {
        const uint32_t lbase = (uint32_t)buf * 8192u;
#pragma unroll
        for (int i = 0; i < 2; ++i) {          // A: 128 rows x 64B = 512 chunks
            int q  = tid + 256 * i;
            int r  = q >> 2;
            int co = (q & 3) * 8;
            uint32_t goff = (uint32_t)(((m0 + r) * DIM + k0 + co) * 2);
            async_ld_b128(asA + lbase + (uint32_t)q * 16u, A, goff);
        }
#pragma unroll
        for (int i = 0; i < 2; ++i) {          // B: 128 rows x 64B = 512 chunks
            int q  = tid + 256 * i;
            int r  = q >> 2;
            int co = (q & 3) * 8;
            uint32_t goff = (uint32_t)(((n0 + r) * DIM + k0 + co) * 2);
            async_ld_b128(asB + lbase + (uint32_t)q * 16u, Bp, goff);
        }
    };

    v8f c[2][4] = {};
    constexpr int NK = DIM / 32;               // 16 K-steps

    issue_tile(0, 0);
    wait_asynccnt0();
    __syncthreads();

    for (int kt = 0; kt < NK; ++kt) {
        const int cur = kt & 1;
        if (kt + 1 < NK) issue_tile((kt + 1) * 32, cur ^ 1);  // prefetch next tile

        const _Float16* Ab = As[cur];
        const _Float16* Bb = Bs[cur];
        v16h a0 = frag_a(Ab + (wm * 32) * 32, 32);
        v16h a1 = frag_a(Ab + (wm * 32 + 16) * 32, 32);
#pragma unroll
        for (int jn = 0; jn < 4; ++jn) {
            v16h bf = frag_b_trans(Bb + (wn * 64 + jn * 16) * 32, 32);
            c[0][jn] = WMMA_F16(a0, bf, c[0][jn]);
            c[1][jn] = WMMA_F16(a1, bf, c[1][jn]);
        }

        wait_asynccnt0();                       // next tile landed
        __syncthreads();                        // all waves done reading `cur`
    }

    const int l  = tid & 31;
    const int ln = l & 15;
    const int mb = (l >> 4) << 3;
    const size_t bbase = (size_t)b * SLQ * SLK;
#pragma unroll
    for (int im = 0; im < 2; ++im)
#pragma unroll
        for (int jn = 0; jn < 4; ++jn)
#pragma unroll
            for (int r = 0; r < 8; ++r) {
                int i = m0 + wm * 32 + im * 16 + mb + r;
                int j = n0 + wn * 64 + jn * 16 + ln;
                size_t idx = bbase + (size_t)i * SLK + j;
                float sc = c[im][jn][r] * INV_TEMP;
                S[idx] = mask[idx] ? -__builtin_inff() : sc;
            }
}

// ---------------------------------------------------------------------------
// Kernel 3: row softmax over attn (in place). One block per (b, i) row.
// ---------------------------------------------------------------------------
__global__ __launch_bounds__(256) void softmax_kernel(float* __restrict__ attn) {
    __shared__ float red[256];
    const int tid = threadIdx.x;
    float* p = attn + (size_t)blockIdx.x * SLK;

    float v[8];
    float mx = -__builtin_inff();
#pragma unroll
    for (int t = 0; t < 8; ++t) {
        v[t] = p[tid + 256 * t];
        mx = fmaxf(mx, v[t]);
    }
    red[tid] = mx;
    __syncthreads();
    for (int s = 128; s > 0; s >>= 1) {
        if (tid < s) red[tid] = fmaxf(red[tid], red[tid + s]);
        __syncthreads();
    }
    mx = red[0];
    __syncthreads();

    const bool dead = (mx == -__builtin_inff());
    float sum = 0.0f;
#pragma unroll
    for (int t = 0; t < 8; ++t) {
        float e = dead ? 0.0f : __expf(v[t] - mx);
        v[t] = e;
        sum += e;
    }
    red[tid] = sum;
    __syncthreads();
    for (int s = 128; s > 0; s >>= 1) {
        if (tid < s) red[tid] += red[tid + s];
        __syncthreads();
    }
    sum = red[0];

    const float inv = dead ? 0.0f : 1.0f / sum;
#pragma unroll
    for (int t = 0; t < 8; ++t) p[tid + 256 * t] = v[t] * inv;
}

// ---------------------------------------------------------------------------
// Kernel 4: output GEMM  O[b,i,d] = sum_j attn[b,i,j] * v[b,j,d]
// Block tile 128x128, wave tile 32x64. attn/v are f32 -> packed f16 staging.
// ---------------------------------------------------------------------------
__global__ __launch_bounds__(256) void out_kernel(const float* __restrict__ P,
                                                  const float* __restrict__ V,
                                                  float* __restrict__ O) {
    __shared__ _Float16 As[128 * 32];   // [i][j]
    __shared__ _Float16 Bs[32 * 128];   // [j][d] direct
    const int tid = threadIdx.x;
    const int wid = tid >> 5;
    const int wm = wid >> 1, wn = wid & 1;     // wave tile 32x64
    const int m0 = blockIdx.x * 128;
    const int n0 = blockIdx.y * 128;
    const int b  = blockIdx.z;
    const float* Pb = P + (size_t)b * SLQ * SLK;
    const float* Vb = V + (size_t)b * SLK * DIM;

    v8f c[2][4] = {};

    for (int k0 = 0; k0 < SLK; k0 += 32) {
#pragma unroll
        for (int i = 0; i < 8; ++i) {              // attn tile, f32->f16 pairs
            int u = tid + 256 * i;                 // 16 pairs/row
            int r = u >> 4, c2 = (u & 15) * 2;
            ((h2*)As)[u] = cvt_pk(&Pb[(size_t)(m0 + r) * SLK + k0 + c2]);
        }
#pragma unroll
        for (int i = 0; i < 8; ++i) {              // v tile [k][n], f32->f16 pairs
            int u = tid + 256 * i;                 // 64 pairs/row (128 f16)
            int r = u >> 6, c2 = (u & 63) * 2;
            ((h2*)Bs)[u] = cvt_pk(&Vb[(size_t)(k0 + r) * DIM + n0 + c2]);
        }
        __syncthreads();

        v16h a0 = frag_a(As + (wm * 32) * 32, 32);
        v16h a1 = frag_a(As + (wm * 32 + 16) * 32, 32);
#pragma unroll
        for (int jn = 0; jn < 4; ++jn) {
            v16h bf = frag_b_direct(Bs + wn * 64 + jn * 16, 128);
            c[0][jn] = WMMA_F16(a0, bf, c[0][jn]);
            c[1][jn] = WMMA_F16(a1, bf, c[1][jn]);
        }
        __syncthreads();
    }

    const int l  = tid & 31;
    const int ln = l & 15;
    const int mb = (l >> 4) << 3;
#pragma unroll
    for (int im = 0; im < 2; ++im)
#pragma unroll
        for (int jn = 0; jn < 4; ++jn)
#pragma unroll
            for (int r = 0; r < 8; ++r) {
                int m = m0 + wm * 32 + im * 16 + mb + r;
                int n = n0 + wn * 64 + jn * 16 + ln;
                O[(size_t)b * SLQ * DIM + (size_t)m * DIM + n] = c[im][jn][r];
            }
}

// ---------------------------------------------------------------------------
// Launch: proj(q), proj(k), scores+mask, softmax, attn@v
// d_in order: q, k, v, mask, Wq, bq, Wk, bk
// d_out: [output (B*LQ*DV f32)] ++ [safe_attn (B*LQ*LK f32)]
// d_ws: qp_f16 (32 MB) ++ kp_f16 (32 MB)
// ---------------------------------------------------------------------------
extern "C" void kernel_launch(void* const* d_in, const int* in_sizes, int n_in,
                              void* d_out, int out_size, void* d_ws, size_t ws_size,
                              hipStream_t stream) {
    const float* q  = (const float*)d_in[0];
    const float* k  = (const float*)d_in[1];
    const float* v  = (const float*)d_in[2];
    const unsigned char* mask = (const unsigned char*)d_in[3];  // jnp bool = 1 byte
    const float* Wq = (const float*)d_in[4];
    const float* bq = (const float*)d_in[5];
    const float* Wk = (const float*)d_in[6];
    const float* bk = (const float*)d_in[7];

    float* out  = (float*)d_out;
    float* attn = out + (size_t)BATCH * SLQ * DIM;

    _Float16* qph = (_Float16*)d_ws;
    _Float16* kph = qph + (size_t)BATCH * SLQ * DIM;

    dim3 blk(256);
    proj_kernel<<<dim3((BATCH * SLQ) / 128, DIM / 64), blk, 0, stream>>>(q, Wq, bq, qph);
    proj_kernel<<<dim3((BATCH * SLK) / 128, DIM / 64), blk, 0, stream>>>(k, Wk, bk, kph);
    scores_kernel<<<dim3(SLQ / 128, SLK / 128, BATCH), blk, 0, stream>>>(qph, kph, mask, attn);
    softmax_kernel<<<dim3(BATCH * SLQ), blk, 0, stream>>>(attn);
    out_kernel<<<dim3(SLQ / 128, DIM / 128, BATCH), blk, 0, stream>>>(attn, v, out);
}
